// EdgeGuidance_59407987638782
// MI455X (gfx1250) — compile-verified, matrix operand order checked by reading, not run on hardware
//
#include <hip/hip_runtime.h>
#include <math.h>

// ---------------------------------------------------------------------------
// CDNA5 / gfx1250 implementation.
// Convs run as implicit GEMM on V_WMMA_F32_16X16X32_F16 with double-buffered
// LDS tiles: global loads for step s+1 are issued before the WMMAs of step s,
// staged through register arrays and written to LDS as packed b128 stores.
// Edge-perception runs wave-per-pixel on an NHWC-staged copy with shfl
// butterfly dot reductions.
// ---------------------------------------------------------------------------

typedef __attribute__((ext_vector_type(16))) _Float16 v16h;
typedef __attribute__((ext_vector_type(8)))  _Float16 v8h;
typedef __attribute__((ext_vector_type(8)))  float    v8f;

#define TCOUT 128   // output channels per block tile (== full Cout)
#define TN     64   // spatial positions per block tile

// ---------------------------------------------------------------------------
// conv3x3 + bias + BN + ReLU, implicit GEMM, double-buffered LDS pipeline.
// Input may be a virtual concat of two NCHW tensors (inA: ch [0,cinA),
// inB: ch [cinA,cinA+cinB)). Output: NCHW, 128 channels, f32.
// Block: 256 threads (8 waves). Each wave: one 16-row M stripe x 64 columns.
// ---------------------------------------------------------------------------
__global__ __launch_bounds__(256) void conv3_bn_relu_wmma(
    const float* __restrict__ inA, int cinA,
    const float* __restrict__ inB, int cinB,
    const float* __restrict__ wgt,    // OIHW f32, O=128, I=cinA+cinB
    const float* __restrict__ cbias,
    const float* __restrict__ bng, const float* __restrict__ bnb,
    const float* __restrict__ bnm, const float* __restrict__ bnv,
    float* __restrict__ out, int B, int H, int W)
{
  const int cin = cinA + cinB;
  const int nc  = cin >> 5;           // K chunks of 32 per (ky,kx)
  const int HW  = H * W;

  __shared__ __align__(32) _Float16 lA[2][TCOUT * 32];  // [m][k]  2 x 8 KB
  __shared__ __align__(32) _Float16 lB[2][TN * 32];     // [n][k]  2 x 4 KB

  const int tid  = threadIdx.x;
  const int lane = tid & 31;
  const int wvid = tid >> 5;              // wave 0..7
  const int n0g  = blockIdx.x * TN;       // first flattened pixel of tile

  // weight loader: thread owns one M row, 16 contiguous K -> one v16h store
  const int wm  = tid & 127;
  const int wkb = (tid >> 7) << 4;        // 0 or 16

  // activation loader: thread owns one N column, 8 contiguous K -> one v8h store
  const int an  = tid & 63;
  const int akb = (tid >> 6) << 3;        // 0,8,16,24
  const int p   = n0g + an;
  const int ax  = p % W;
  const int ay  = (p / W) % H;
  const int ab  = p / HW;

  v8f acc0 = {}; v8f acc1 = {}; v8f acc2 = {}; v8f acc3 = {};

  // ---- tile load (global -> regs) and commit (regs -> LDS, packed) --------
  auto load_tile = [&](int s, float (&wr)[16], float (&ar)[8]) {
    const int kyx = s / nc;                     // 0..8, koff == kyx
    const int c0  = (s - kyx * nc) << 5;
    const int ky  = kyx / 3 - 1;
    const int kx  = kyx % 3 - 1;
    // weights: 16 consecutive channels of row wm, kernel tap kyx
    const float* wp = wgt + (size_t)wm * cin * 9 + (size_t)(c0 + wkb) * 9 + kyx;
#pragma unroll
    for (int i = 0; i < 16; ++i) wr[i] = wp[(size_t)i * 9];
    // activations: 8 consecutive channels at shifted pixel, zero-padded halo
    const int yy = ay + ky;
    const int xx = ax + kx;
    const bool inb = (yy >= 0) && (yy < H) && (xx >= 0) && (xx < W);
#pragma unroll
    for (int i = 0; i < 8; ++i) {
      const int c = c0 + akb + i;
      float v = 0.0f;
      if (inb) {
        if (c < cinA)
          v = inA[(((size_t)ab * cinA + c) * H + yy) * W + xx];
        else
          v = inB[(((size_t)ab * cinB + (c - cinA)) * H + yy) * W + xx];
      }
      ar[i] = v;
    }
  };
  auto store_tile = [&](int buf, const float (&wr)[16], const float (&ar)[8]) {
    v16h wh;
#pragma unroll
    for (int i = 0; i < 16; ++i) wh[i] = (_Float16)wr[i];
    *(v16h*)&lA[buf][wm * 32 + wkb] = wh;       // 32 B packed DS store
    v8h ah;
#pragma unroll
    for (int i = 0; i < 8; ++i) ah[i] = (_Float16)ar[i];
    *(v8h*)&lB[buf][an * 32 + akb] = ah;        // 16 B packed DS store
  };

  // ---- software-pipelined main loop ---------------------------------------
  const int steps = 9 * nc;
  {
    float wr[16], ar[8];
    load_tile(0, wr, ar);
    store_tile(0, wr, ar);
  }
  int cur = 0;
  const int row = (wvid << 4) + (lane & 15);
  const int kh  = (lane >> 4) << 4;       // lanes 0-15: K 0..15, 16-31: K 16..31
  const int cl  = lane & 15;

  for (int s = 0; s < steps; ++s) {
    __syncthreads();                      // tile[cur] visible; tile[cur^1] free
    float wr[16], ar[8];
    const bool more = (s + 1) < steps;
    if (more) load_tile(s + 1, wr, ar);   // global loads in flight during WMMA

    const v16h a = *(const v16h*)&lA[cur][row * 32 + kh];
    v16h b;
    b = *(const v16h*)&lB[cur][(cl +  0) * 32 + kh];
    acc0 = __builtin_amdgcn_wmma_f32_16x16x32_f16(false, a, false, b, (short)0, acc0, false, false);
    b = *(const v16h*)&lB[cur][(cl + 16) * 32 + kh];
    acc1 = __builtin_amdgcn_wmma_f32_16x16x32_f16(false, a, false, b, (short)0, acc1, false, false);
    b = *(const v16h*)&lB[cur][(cl + 32) * 32 + kh];
    acc2 = __builtin_amdgcn_wmma_f32_16x16x32_f16(false, a, false, b, (short)0, acc2, false, false);
    b = *(const v16h*)&lB[cur][(cl + 48) * 32 + kh];
    acc3 = __builtin_amdgcn_wmma_f32_16x16x32_f16(false, a, false, b, (short)0, acc3, false, false);

    if (more) store_tile(cur ^ 1, wr, ar);
    cur ^= 1;
  }

  // ---- epilogue: fold conv bias + BN, ReLU, scatter NCHW ----
  // C/D layout: VGPR r -> M = r (lanes 0-15) / M = 8+r (lanes 16-31), N = lane&15
  const int m0 = (wvid << 4) + ((lane >> 4) << 3);
  v8f accs[4] = {acc0, acc1, acc2, acc3};
#pragma unroll
  for (int nt = 0; nt < 4; ++nt) {
    const int pp = n0g + nt * 16 + cl;
    const int ox = pp % W;
    const int oy = (pp / W) % H;
    const int ob = pp / HW;
#pragma unroll
    for (int r = 0; r < 8; ++r) {
      const int ch = m0 + r;
      const float sc = bng[ch] * rsqrtf(bnv[ch] + 1e-5f);
      float v = accs[nt][r] * sc + (cbias[ch] - bnm[ch]) * sc + bnb[ch];
      v = fmaxf(v, 0.0f);
      out[(((size_t)ob * TCOUT + ch) * H + oy) * W + ox] = v;
    }
  }
}

// ---------------------------------------------------------------------------
// bilinear x2 resize (half-pixel centers, edge clamp) on NCHW f32
// ---------------------------------------------------------------------------
__global__ __launch_bounds__(256) void resize_bilinear(
    const float* __restrict__ in, float* __restrict__ out,
    int BC, int Hi, int Wi, int Ho, int Wo)
{
  size_t i = (size_t)blockIdx.x * blockDim.x + threadIdx.x;
  const size_t total = (size_t)BC * Ho * Wo;
  if (i >= total) return;
  const int xo = (int)(i % Wo);
  const int yo = (int)((i / Wo) % Ho);
  const int bc = (int)(i / ((size_t)Wo * Ho));
  const float sy = (float)Hi / (float)Ho;
  const float sx = (float)Wi / (float)Wo;
  float fy = ((float)yo + 0.5f) * sy - 0.5f;
  float fx = ((float)xo + 0.5f) * sx - 0.5f;
  int y0 = (int)floorf(fy); float wy = fy - (float)y0;
  int x0 = (int)floorf(fx); float wx = fx - (float)x0;
  int y0c = min(max(y0, 0), Hi - 1), y1c = min(max(y0 + 1, 0), Hi - 1);
  int x0c = min(max(x0, 0), Wi - 1), x1c = min(max(x0 + 1, 0), Wi - 1);
  const float* pb = in + (size_t)bc * Hi * Wi;
  float v = (1.f - wy) * ((1.f - wx) * pb[(size_t)y0c * Wi + x0c] + wx * pb[(size_t)y0c * Wi + x1c])
          +        wy  * ((1.f - wx) * pb[(size_t)y1c * Wi + x0c] + wx * pb[(size_t)y1c * Wi + x1c]);
  out[i] = v;
}

// ---------------------------------------------------------------------------
// global max / mean pool over H*W: one block per (b,c)
// ---------------------------------------------------------------------------
__global__ __launch_bounds__(256) void pool_max_kernel(
    const float* __restrict__ in, float* __restrict__ out, int HW)
{
  const int bc = blockIdx.x;
  const float* p = in + (size_t)bc * HW;
  float v = -3.4e38f;
  for (int i = threadIdx.x; i < HW; i += 256) v = fmaxf(v, p[i]);
  __shared__ float red[256];
  red[threadIdx.x] = v; __syncthreads();
  for (int s = 128; s; s >>= 1) {
    if (threadIdx.x < s) red[threadIdx.x] = fmaxf(red[threadIdx.x], red[threadIdx.x + s]);
    __syncthreads();
  }
  if (threadIdx.x == 0) out[bc] = red[0];
}

__global__ __launch_bounds__(256) void pool_mean_kernel(
    const float* __restrict__ in, float* __restrict__ out, int HW)
{
  const int bc = blockIdx.x;
  const float* p = in + (size_t)bc * HW;
  float v = 0.f;
  for (int i = threadIdx.x; i < HW; i += 256) v += p[i];
  __shared__ float red[256];
  red[threadIdx.x] = v; __syncthreads();
  for (int s = 128; s; s >>= 1) {
    if (threadIdx.x < s) red[threadIdx.x] += red[threadIdx.x + s];
    __syncthreads();
  }
  if (threadIdx.x == 0) out[bc] = red[0] / (float)HW;
}

// ---------------------------------------------------------------------------
// NCHW -> NHWC (C=128) staging for the per-pixel channel dots
// ---------------------------------------------------------------------------
__global__ __launch_bounds__(256) void nchw_to_nhwc128(
    const float* __restrict__ in, float* __restrict__ out, int B, int HW)
{
  size_t i = (size_t)blockIdx.x * blockDim.x + threadIdx.x;
  const size_t total = (size_t)B * 128 * HW;
  if (i >= total) return;
  const int x = (int)(i % HW);
  const int c = (int)((i / HW) % 128);
  const int b = (int)(i / ((size_t)HW * 128));
  out[((size_t)b * HW + x) * 128 + c] = in[i];
}

// per-pixel squared L2 norm over C=128 (wave per pixel, lane = 4 channels)
__global__ __launch_bounds__(256) void norm2_kernel(
    const float* __restrict__ xt, float* __restrict__ n2, int P)
{
  const int wv = (int)(((size_t)blockIdx.x * blockDim.x + threadIdx.x) >> 5);
  const int lane = threadIdx.x & 31;
  if (wv >= P) return;
  float4 f = ((const float4*)(xt + (size_t)wv * 128))[lane];
  float s = f.x * f.x + f.y * f.y + f.z * f.z + f.w * f.w;
  for (int o = 16; o; o >>= 1) s += __shfl_xor(s, o);
  if (lane == 0) n2[wv] = s;
}

// pass A: 9x9 cosine-sim threshold count -> m = sqrt(count)   (wave/pixel)
__global__ __launch_bounds__(256) void ep_count9_kernel(
    const float* __restrict__ xt, const float* __restrict__ n2,
    const float* __restrict__ t1p, float* __restrict__ mout,
    int B, int H, int W)
{
  const int P = B * H * W;
  const int wv = (int)(((size_t)blockIdx.x * blockDim.x + threadIdx.x) >> 5);
  const int lane = threadIdx.x & 31;
  if (wv >= P) return;
  const int x = wv % W, y = (wv / W) % H, b = wv / (W * H);
  const float4 cv = ((const float4*)(xt + (size_t)wv * 128))[lane];
  const float cn2 = n2[wv];
  const float t1 = *t1p;
  int cnt = 0;
  for (int dy = -4; dy <= 4; ++dy) {
    for (int dx = -4; dx <= 4; ++dx) {
      if (dy == 0 && dx == 0) continue;
      const int yy = y + dy, xx = x + dx;
      float sim = 0.0f;  // zero-padded neighbor -> sim = 0
      if (yy >= 0 && yy < H && xx >= 0 && xx < W) {
        const int q = (b * H + yy) * W + xx;
        const float4 nv = ((const float4*)(xt + (size_t)q * 128))[lane];
        float d = cv.x * nv.x + cv.y * nv.y + cv.z * nv.z + cv.w * nv.w;
        for (int o = 16; o; o >>= 1) d += __shfl_xor(d, o);
        sim = d / fmaxf(sqrtf(cn2 * n2[q]), 1e-8f);
      }
      cnt += (sim < t1) ? 1 : 0;
    }
  }
  if (lane == 0) mout[wv] = sqrtf((float)cnt);
}

// pass B: 5x5 sims of xm=(1+m)x via scaled dots/norms; final
// out = x * (1 + m2) * gmp[b,c] + gap[b,c]    (wave/pixel, NCHW scatter)
__global__ __launch_bounds__(256) void ep_final5_kernel(
    const float* __restrict__ xt, const float* __restrict__ n2,
    const float* __restrict__ m, const float* __restrict__ t2p,
    const float* __restrict__ gmp, const float* __restrict__ gap,
    float* __restrict__ out, int B, int H, int W)
{
  const int P = B * H * W;
  const int wv = (int)(((size_t)blockIdx.x * blockDim.x + threadIdx.x) >> 5);
  const int lane = threadIdx.x & 31;
  if (wv >= P) return;
  const int x = wv % W, y = (wv / W) % H, b = wv / (W * H);
  const float4 cv = ((const float4*)(xt + (size_t)wv * 128))[lane];
  const float sc  = 1.0f + m[wv];
  const float cn2 = sc * sc * n2[wv];
  const float t2 = *t2p;
  int cnt = 0;
  for (int dy = -2; dy <= 2; ++dy) {
    for (int dx = -2; dx <= 2; ++dx) {
      if (dy == 0 && dx == 0) continue;
      const int yy = y + dy, xx = x + dx;
      float sim = 0.0f;
      if (yy >= 0 && yy < H && xx >= 0 && xx < W) {
        const int q = (b * H + yy) * W + xx;
        const float4 nv = ((const float4*)(xt + (size_t)q * 128))[lane];
        float d = cv.x * nv.x + cv.y * nv.y + cv.z * nv.z + cv.w * nv.w;
        for (int o = 16; o; o >>= 1) d += __shfl_xor(d, o);
        const float sq = 1.0f + m[q];
        sim = (sc * sq * d) / fmaxf(sqrtf(cn2 * (sq * sq * n2[q])), 1e-8f);
      }
      cnt += (sim < t2) ? 1 : 0;
    }
  }
  const float fac = 1.0f + sqrtf((float)cnt);
  const int cbase = lane * 4;
  const float cvv[4] = {cv.x, cv.y, cv.z, cv.w};
#pragma unroll
  for (int j = 0; j < 4; ++j) {
    const int c = cbase + j;
    out[(((size_t)b * 128 + c) * H + y) * W + x] =
        cvv[j] * fac * gmp[b * 128 + c] + gap[b * 128 + c];
  }
}

// ---------------------------------------------------------------------------
// host orchestration
// ---------------------------------------------------------------------------
extern "C" void kernel_launch(void* const* d_in, const int* in_sizes, int n_in,
                              void* d_out, int out_size, void* d_ws, size_t ws_size,
                              hipStream_t stream)
{
  (void)in_sizes; (void)n_in; (void)out_size; (void)ws_size;
  // inputs: [0..3]=x0..x3; [4+6i..9+6i]=conv{i}_w,conv{i}_b,bn{i}_g,bn{i}_b,
  // bn{i}_m,bn{i}_v for i=0..9; [64..69]=ep1_t1,ep1_t2,ep2_t1,ep2_t2,ep3_t1,ep3_t2
  const float* x0 = (const float*)d_in[0];
  const float* x1 = (const float*)d_in[1];
  const float* x2 = (const float*)d_in[2];
  const float* x3 = (const float*)d_in[3];
  auto PW  = [&](int i) { return (const float*)d_in[4 + i * 6 + 0]; };
  auto PB  = [&](int i) { return (const float*)d_in[4 + i * 6 + 1]; };
  auto BG  = [&](int i) { return (const float*)d_in[4 + i * 6 + 2]; };
  auto BB  = [&](int i) { return (const float*)d_in[4 + i * 6 + 3]; };
  auto BM  = [&](int i) { return (const float*)d_in[4 + i * 6 + 4]; };
  auto BV  = [&](int i) { return (const float*)d_in[4 + i * 6 + 5]; };
  const float* ep_t[6] = {(const float*)d_in[64], (const float*)d_in[65],
                          (const float*)d_in[66], (const float*)d_in[67],
                          (const float*)d_in[68], (const float*)d_in[69]};

  // workspace arena (f32 elems): ~220 MB total
  float* ws = (float*)d_ws;
  const size_t s16 = 8u * 128u * 16u * 16u;    // 262144
  const size_t s32 = s16 * 4;                  // 1048576
  const size_t s64 = s32 * 4;                  // 4194304
  const size_t s128 = s64 * 4;                 // 16777216
  float* OUT0   = ws;                          // s16
  float* OUT0R  = OUT0 + s16;                  // s32
  float* CAT32  = OUT0R + s32;                 // s32
  float* TMP32  = CAT32 + s32;                 // s32
  float* FIN32  = TMP32 + s32;                 // s32
  float* OUT1R  = FIN32 + s32;                 // s64
  float* CAT64  = OUT1R + s64;                 // s64
  float* TMP64  = CAT64 + s64;                 // s64
  float* FIN64  = TMP64 + s64;                 // s64
  float* OUT2R  = FIN64 + s64;                 // s128 (aliased as NHWC staging XT)
  float* CAT128 = OUT2R + s128;                // s128
  float* XT     = OUT2R;                       // alias: dead at EP time at each scale
  float* N2B    = CAT128 + s128;               // 131072 (max P)
  float* MB     = N2B + 131072;                // 131072
  float* GMP    = MB + 131072;                 // 1024
  float* GAP    = GMP + 1024;                  // 1024
  float* OUTF   = (float*)d_out;               // also conv8/conv9 staging

  const int B = 8;

  auto conv = [&](const float* iA, int cA, const float* iB, int cB, int i,
                  float* o, int H, int W) {
    const int nblk = (B * H * W) / TN;
    conv3_bn_relu_wmma<<<nblk, 256, 0, stream>>>(iA, cA, iB, cB, PW(i), PB(i),
                                                 BG(i), BB(i), BM(i), BV(i),
                                                 o, B, H, W);
  };
  auto resize2x = [&](const float* in, float* o, int Hi, int Wi) {
    const size_t tot = (size_t)B * 128 * (Hi * 2) * (Wi * 2);
    resize_bilinear<<<(unsigned)((tot + 255) / 256), 256, 0, stream>>>(
        in, o, B * 128, Hi, Wi, Hi * 2, Wi * 2);
  };
  auto edge = [&](const float* cat, const float* t1, const float* t2,
                  float* o, int H, int W) {
    const int HW = H * W, P = B * HW;
    const size_t tot = (size_t)B * 128 * HW;
    nchw_to_nhwc128<<<(unsigned)((tot + 255) / 256), 256, 0, stream>>>(cat, XT, B, HW);
    norm2_kernel<<<(P + 7) / 8, 256, 0, stream>>>(XT, N2B, P);
    ep_count9_kernel<<<(P + 7) / 8, 256, 0, stream>>>(XT, N2B, t1, MB, B, H, W);
    ep_final5_kernel<<<(P + 7) / 8, 256, 0, stream>>>(XT, N2B, MB, t2, GMP, GAP, o, B, H, W);
  };

  // ---- scale 16 -> 32 ----
  conv(x0, 128, x0, 0, 0, OUT0, 16, 16);                 // out0 = cbr0(x0)
  resize2x(OUT0, OUT0R, 16, 16);                         // out0 -> 32x32
  conv(x1, 128, OUT0R, 128, 1, CAT32, 32, 32);           // cbr1(cat(x1,out0))
  conv(x1, 128, x1, 0, 4, TMP32, 32, 32);                // cbr4(x1)
  pool_max_kernel<<<1024, 256, 0, stream>>>(TMP32, GMP, 32 * 32);
  conv(OUT0R, 128, OUT0R, 0, 5, TMP32, 32, 32);          // cbr5(out0)
  pool_mean_kernel<<<1024, 256, 0, stream>>>(TMP32, GAP, 32 * 32);
  edge(CAT32, ep_t[0], ep_t[1], FIN32, 32, 32);          // out0 final

  // ---- scale 32 -> 64 ----
  resize2x(FIN32, OUT1R, 32, 32);
  conv(x2, 128, OUT1R, 128, 2, CAT64, 64, 64);
  conv(x2, 128, x2, 0, 6, TMP64, 64, 64);
  pool_max_kernel<<<1024, 256, 0, stream>>>(TMP64, GMP, 64 * 64);
  conv(OUT1R, 128, OUT1R, 0, 7, TMP64, 64, 64);
  pool_mean_kernel<<<1024, 256, 0, stream>>>(TMP64, GAP, 64 * 64);
  edge(CAT64, ep_t[2], ep_t[3], FIN64, 64, 64);          // out1 final

  // ---- scale 64 -> 128 ----
  resize2x(FIN64, OUT2R, 64, 64);
  conv(x3, 128, OUT2R, 128, 3, CAT128, 128, 128);
  conv(x3, 128, x3, 0, 8, OUTF, 128, 128);               // stage in d_out
  pool_max_kernel<<<1024, 256, 0, stream>>>(OUTF, GMP, 128 * 128);
  conv(OUT2R, 128, OUT2R, 0, 9, OUTF, 128, 128);         // stage in d_out
  pool_mean_kernel<<<1024, 256, 0, stream>>>(OUTF, GAP, 128 * 128);
  edge(CAT128, ep_t[4], ep_t[5], OUTF, 128, 128);        // final output
}
